// FullODENet_23622320128057
// MI455X (gfx1250) — matrix operand order repaired and checked by reading, not compiled, not run on hardware
//
#include <hip/hip_runtime.h>
#include <hip/hip_bf16.h>
#include <cstdint>
#include <cstddef>

typedef __bf16 bf16;
typedef __attribute__((ext_vector_type(8)))  bf16  v8bf;
typedef __attribute__((ext_vector_type(16))) bf16  v16bf;
typedef __attribute__((ext_vector_type(8)))  float v8f;

// ---------------------------------------------------------------------------
// WMMA GEMM: C[M,N] (f32) = A[M,K](bf16) x B[K,N](bf16, stored transposed as
// Bt[N,K]) (+ C_in if accum) (+ bias[n]).  K % 32 == 0, M % 16 == 0,
// N == 16*NT.  One wave computes a 16 x (16*NT) strip: the A fragment is
// loaded ONCE per K-step and reused across NT WMMAs (A is the big im2col
// stream; B is tiny and stays in L0/L2).  A/B fragments use k = 16*(lane/16) + s
// in both A and B (wave32).
// ---------------------------------------------------------------------------
template <int NT>
__global__ __launch_bounds__(256)
void wmma_gemm_bf16(const bf16* __restrict__ A,
                    const bf16* __restrict__ Bt,
                    float* __restrict__ C,
                    const float* __restrict__ bias,
                    int M, int N, int K, int accum)
{
    const int wave = threadIdx.x >> 5;
    const int lane = threadIdx.x & 31;
    const int m0 = (blockIdx.x * 8 + wave) * 16;
    if (m0 >= M) return;

    const int col = lane & 15;   // N column within tile (and A row offset)
    const int hi  = lane >> 4;   // half-wave selector

    v8f c[NT];
#pragma unroll
    for (int j = 0; j < NT; ++j) {
        if (accum) {
#pragma unroll
            for (int r = 0; r < 8; ++r)
                c[j][r] = C[(size_t)(m0 + r + hi * 8) * N + (j * 16 + col)];
        } else {
#pragma unroll
            for (int r = 0; r < 8; ++r) c[j][r] = 0.f;
        }
    }

    // A: lane holds row (m0+col); lanes 0-15 cover K 0..15, lanes
    // 16-31 cover K 16..31 within each 32-wide K tile.
    const bf16* Arow = A + (size_t)(m0 + col) * K + hi * 16;
    // Bt: lane holds output column (j*16+col); lanes 0-15 cover K 0..15,
    // lanes 16-31 cover K 16..31 (contiguous since Bt is [N,K]).
    const bf16* Brow[NT];
#pragma unroll
    for (int j = 0; j < NT; ++j)
        Brow[j] = Bt + (size_t)(j * 16 + col) * K + hi * 16;

    for (int k = 0; k < K; k += 32) {
        v8bf a_lo = *(const v8bf*)(Arow + k);
        v8bf a_hi = *(const v8bf*)(Arow + k + 8);
        v16bf a = __builtin_shufflevector(a_lo, a_hi,
                                          0,1,2,3,4,5,6,7,8,9,10,11,12,13,14,15);
#pragma unroll
        for (int j = 0; j < NT; ++j) {
            v16bf b = *(const v16bf*)(Brow[j] + k);
            c[j] = __builtin_amdgcn_wmma_f32_16x16x32_bf16(false, a, false, b,
                                                           (short)0, c[j],
                                                           false, false);
        }
    }

#pragma unroll
    for (int j = 0; j < NT; ++j) {
        const float bv = bias ? bias[j * 16 + col] : 0.f;
#pragma unroll
        for (int r = 0; r < 8; ++r)
            C[(size_t)(m0 + r + hi * 8) * N + (j * 16 + col)] = c[j][r] + bv;
    }
}

// ---------------------------------------------------------------------------
// im2col (NHWC, SAME padding) with fp32 -> bf16 conversion.  Optionally
// prepends a constant-t input channel (ConcatConv2D), padded K -> Kpad.
// dst is [B*Ho*Wo, Kpad], k = (kh*ks + kw) * Cin + cin, cin==0 is t-channel.
// ---------------------------------------------------------------------------
__global__ __launch_bounds__(256)
void im2col_bf16(const float* __restrict__ src, bf16* __restrict__ dst,
                 int Bn, int Hs, int Ws, int Cs,
                 int Ho, int Wo, int ksize, int stride, int pad,
                 int Kpad, int has_t, float tval)
{
    const int Cin   = Cs + has_t;
    const int Kreal = ksize * ksize * Cin;
    const size_t total    = (size_t)Bn * Ho * Wo * Kpad;
    const size_t stride_t = (size_t)gridDim.x * blockDim.x;
    for (size_t i = (size_t)blockIdx.x * blockDim.x + threadIdx.x;
         i < total; i += stride_t) {
        const int    k = (int)(i % Kpad);
        const size_t m = i / Kpad;
        float v = 0.f;
        if (k < Kreal) {
            const int ow  = (int)(m % Wo);
            const int oh  = (int)((m / Wo) % Ho);
            const int n   = (int)(m / ((size_t)Wo * Ho));
            const int cin = k % Cin;
            const int kk  = k / Cin;
            const int kw  = kk % ksize;
            const int kh  = kk / ksize;
            const int ih  = oh * stride + kh - pad;
            const int iw  = ow * stride + kw - pad;
            if (ih >= 0 && ih < Hs && iw >= 0 && iw < Ws) {
                if (has_t && cin == 0) v = tval;
                else v = src[(((size_t)n * Hs + ih) * Ws + iw) * Cs + (cin - has_t)];
            }
        }
        dst[i] = (bf16)v;
    }
}

// ---------------------------------------------------------------------------
// Pack HWIO fp32 weights (flat [K, O]) into bf16 Bt [Npad, Kpad] (zero pad).
// ---------------------------------------------------------------------------
__global__ __launch_bounds__(256)
void pack_w_bf16(const float* __restrict__ w, bf16* __restrict__ bt,
                 int K, int O, int Kpad, int Npad)
{
    const size_t total = (size_t)Npad * Kpad;
    const size_t st = (size_t)gridDim.x * blockDim.x;
    for (size_t i = (size_t)blockIdx.x * blockDim.x + threadIdx.x; i < total; i += st) {
        const int k = (int)(i % Kpad);
        const int o = (int)(i / Kpad);
        float v = (k < K && o < O) ? w[(size_t)k * O + o] : 0.f;
        bt[i] = (bf16)v;
    }
}

// ---------------------------------------------------------------------------
// GroupNorm(groups == channels) == per-(n,c) mean/var over spatial dims.
// ---------------------------------------------------------------------------
__global__ __launch_bounds__(256)
void gn_stats(const float* __restrict__ x, float* __restrict__ mean,
              float* __restrict__ rstd, int S, int C)
{
    const int n = blockIdx.x / C;
    const int c = blockIdx.x % C;
    const float* p = x + (size_t)n * S * C + c;
    float s = 0.f, s2 = 0.f;
    for (int i = threadIdx.x; i < S; i += blockDim.x) {
        const float v = p[(size_t)i * C];
        s += v; s2 += v * v;
    }
    __shared__ float sh[256];
    __shared__ float sh2[256];
    sh[threadIdx.x] = s; sh2[threadIdx.x] = s2;
    __syncthreads();
    for (int off = 128; off > 0; off >>= 1) {
        if (threadIdx.x < off) {
            sh[threadIdx.x]  += sh[threadIdx.x + off];
            sh2[threadIdx.x] += sh2[threadIdx.x + off];
        }
        __syncthreads();
    }
    if (threadIdx.x == 0) {
        const float mu  = sh[0] / (float)S;
        const float var = sh2[0] / (float)S - mu * mu;
        mean[blockIdx.x] = mu;
        rstd[blockIdx.x] = rsqrtf(var + 1e-6f);
    }
}

__global__ __launch_bounds__(256)
void gn_apply(const float* __restrict__ x, float* __restrict__ y,
              const float* __restrict__ mean, const float* __restrict__ rstd,
              const float* __restrict__ scale, const float* __restrict__ bias,
              int S, int C, int relu, size_t total)
{
    const size_t st = (size_t)gridDim.x * blockDim.x;
    for (size_t i = (size_t)blockIdx.x * blockDim.x + threadIdx.x;
         i < total; i += st) {
        const int    c   = (int)(i % C);
        const size_t rem = i / C;
        const int    n   = (int)(rem / S);
        const int    idx = n * C + c;
        float v = (x[i] - mean[idx]) * rstd[idx] * scale[c] + bias[c];
        if (relu) v = fmaxf(v, 0.f);
        y[i] = v;
    }
}

// ---------------------------------------------------------------------------
// out[i] = sum_j c_j * p_j[i]   (null p_j skipped)  -- dopri5 state combine
// ---------------------------------------------------------------------------
__global__ __launch_bounds__(256)
void wsum7(float* __restrict__ out, size_t n,
           const float* p0, float c0, const float* p1, float c1,
           const float* p2, float c2, const float* p3, float c3,
           const float* p4, float c4, const float* p5, float c5,
           const float* p6, float c6)
{
    const size_t st = (size_t)gridDim.x * blockDim.x;
    for (size_t i = (size_t)blockIdx.x * blockDim.x + threadIdx.x; i < n; i += st) {
        float a = 0.f;
        if (p0) a += c0 * p0[i];
        if (p1) a += c1 * p1[i];
        if (p2) a += c2 * p2[i];
        if (p3) a += c3 * p3[i];
        if (p4) a += c4 * p4[i];
        if (p5) a += c5 * p5[i];
        if (p6) a += c6 * p6[i];
        out[i] = a;
    }
}

__global__ __launch_bounds__(256)
void cast_f32_bf16(const float* __restrict__ x, bf16* __restrict__ y, size_t n)
{
    const size_t st = (size_t)gridDim.x * blockDim.x;
    for (size_t i = (size_t)blockIdx.x * blockDim.x + threadIdx.x; i < n; i += st)
        y[i] = (bf16)x[i];
}

__global__ void logsoftmax10(const float* __restrict__ Cd,
                             const float* __restrict__ bias,
                             float* __restrict__ out, int Bn)
{
    const int n = blockIdx.x * blockDim.x + threadIdx.x;
    if (n >= Bn) return;
    float v[10];
    float mx = -1e30f;
    for (int i = 0; i < 10; ++i) {
        v[i] = Cd[n * 16 + i] + bias[i];
        mx = fmaxf(mx, v[i]);
    }
    float s = 0.f;
    for (int i = 0; i < 10; ++i) s += expf(v[i] - mx);
    const float l = logf(s) + mx;
    for (int i = 0; i < 10; ++i) out[n * 10 + i] = v[i] - l;
}

// ============================ host-side helpers ============================

static void gemm(hipStream_t s, const bf16* A, const bf16* Bt, float* C,
                 const float* bias, int M, int N, int K, int accum)
{
    dim3 g((M + 127) / 128);
    if (N == 64)
        wmma_gemm_bf16<4><<<g, 256, 0, s>>>(A, Bt, C, bias, M, N, K, accum);
    else // N == 16
        wmma_gemm_bf16<1><<<g, 256, 0, s>>>(A, Bt, C, bias, M, N, K, accum);
}

static void im2col(hipStream_t s, const float* src, bf16* dst, int Bn, int Hs,
                   int Ws, int Cs, int Ho, int Wo, int ks, int stride, int pad,
                   int Kpad, int has_t, float t)
{
    size_t total = (size_t)Bn * Ho * Wo * Kpad;
    int blocks = (int)((total + 255) / 256);
    im2col_bf16<<<blocks, 256, 0, s>>>(src, dst, Bn, Hs, Ws, Cs, Ho, Wo, ks,
                                       stride, pad, Kpad, has_t, t);
}

static void pack(hipStream_t s, const float* w, bf16* bt, int K, int O,
                 int Kpad, int Npad)
{
    size_t total = (size_t)Npad * Kpad;
    pack_w_bf16<<<(int)((total + 255) / 256), 256, 0, s>>>(w, bt, K, O, Kpad, Npad);
}

static void gn_apply_launch(hipStream_t s, const float* x, float* y,
                            const float* mean, const float* rstd,
                            const float* sc, const float* bi, int Bn, int S,
                            int C, int relu)
{
    size_t total = (size_t)Bn * S * C;
    gn_apply<<<(int)((total + 255) / 256), 256, 0, s>>>(x, y, mean, rstd, sc, bi,
                                                        S, C, relu, total);
}

static void gn(hipStream_t s, const float* x, float* y, const float* sc,
               const float* bi, float* mean, float* rstd, int Bn, int S, int C,
               int relu)
{
    gn_stats<<<Bn * C, 256, 0, s>>>(x, mean, rstd, S, C);
    gn_apply_launch(s, x, y, mean, rstd, sc, bi, Bn, S, C, relu);
}

static void wsum(hipStream_t s, float* out, size_t n,
                 const float* p0, float c0,
                 const float* p1 = nullptr, float c1 = 0.f,
                 const float* p2 = nullptr, float c2 = 0.f,
                 const float* p3 = nullptr, float c3 = 0.f,
                 const float* p4 = nullptr, float c4 = 0.f,
                 const float* p5 = nullptr, float c5 = 0.f,
                 const float* p6 = nullptr, float c6 = 0.f)
{
    wsum7<<<(int)((n + 255) / 256), 256, 0, s>>>(out, n, p0, c0, p1, c1, p2, c2,
                                                 p3, c3, p4, c4, p5, c5, p6, c6);
}

struct OdeCtx {
    hipStream_t s;
    const float *gn1_s, *gn1_b, *cc1_b, *gn2_s, *gn2_b, *cc2_b, *gn3_s, *gn3_b;
    const bf16 *Wcc1, *Wcc2;
    bf16* A;
    float *o1, *o2, *mean, *rstd;
};

// ode_func: GN->relu->ConcatConv->GN->relu->ConcatConv->GN  on [128,16,16,64]
static void ode_f(const OdeCtx& c, const float* y, float t, float* kout)
{
    gn(c.s, y, c.o1, c.gn1_s, c.gn1_b, c.mean, c.rstd, 128, 256, 64, 1);
    im2col(c.s, c.o1, c.A, 128, 16, 16, 64, 16, 16, 3, 1, 1, 608, 1, t);
    gemm(c.s, c.A, c.Wcc1, c.o2, c.cc1_b, 32768, 64, 608, 0);
    gn(c.s, c.o2, c.o1, c.gn2_s, c.gn2_b, c.mean, c.rstd, 128, 256, 64, 1);
    im2col(c.s, c.o1, c.A, 128, 16, 16, 64, 16, 16, 3, 1, 1, 608, 1, t);
    gemm(c.s, c.A, c.Wcc2, c.o2, c.cc2_b, 32768, 64, 608, 0);
    gn(c.s, c.o2, kout, c.gn3_s, c.gn3_b, c.mean, c.rstd, 128, 256, 64, 0);
}

// ================================ pipeline =================================

extern "C" void kernel_launch(void* const* d_in, const int* in_sizes, int n_in,
                              void* d_out, int out_size, void* d_ws, size_t ws_size,
                              hipStream_t stream)
{
    (void)in_sizes; (void)n_in; (void)out_size; (void)ws_size;

    // ---- inputs (setup_inputs() dict order, depth-first) ----
    const float* x_in    = (const float*)d_in[0];   // [128,64,64,3]
    const float* conv1_w = (const float*)d_in[1];   // [3,3,3,64]
    const float* conv1_b = (const float*)d_in[2];
    const float* d1_gn_s = (const float*)d_in[3];
    const float* d1_gn_b = (const float*)d_in[4];
    const float* d1_cw   = (const float*)d_in[5];   // [3,3,64,64]
    const float* d1_cb   = (const float*)d_in[6];
    const float* d1_sw   = (const float*)d_in[7];   // [1,1,64,64]
    const float* d1_sb   = (const float*)d_in[8];
    const float* d2_gn_s = (const float*)d_in[9];
    const float* d2_gn_b = (const float*)d_in[10];
    const float* d2_cw   = (const float*)d_in[11];
    const float* d2_cb   = (const float*)d_in[12];
    const float* d2_sw   = (const float*)d_in[13];
    const float* d2_sb   = (const float*)d_in[14];
    const float* gn1_s   = (const float*)d_in[15];
    const float* gn1_b   = (const float*)d_in[16];
    const float* cc1_w   = (const float*)d_in[17];  // [3,3,65,64]
    const float* cc1_b   = (const float*)d_in[18];
    const float* gn2_s   = (const float*)d_in[19];
    const float* gn2_b   = (const float*)d_in[20];
    const float* cc2_w   = (const float*)d_in[21];
    const float* cc2_b   = (const float*)d_in[22];
    const float* gn3_s   = (const float*)d_in[23];
    const float* gn3_b   = (const float*)d_in[24];
    const float* gno_s   = (const float*)d_in[25];
    const float* gno_b   = (const float*)d_in[26];
    const float* dw      = (const float*)d_in[27];  // [16384,10]
    const float* db      = (const float*)d_in[28];

    // ---- workspace bump allocator ----
    size_t cur = 0;
    auto alloc = [&](size_t bytes) -> void* {
        void* p = (char*)d_ws + cur;
        cur += (bytes + 255) & ~(size_t)255;
        return p;
    };

    const int M1  = 128 * 64 * 64;  // 524288
    const int Md1 = 128 * 32 * 32;  // 131072
    const int Md2 = 128 * 16 * 16;  // 32768
    const size_t NE = (size_t)Md2 * 64;

    float* big0 = (float*)alloc((size_t)M1 * 64 * 4);   // conv1 out [128,64,64,64]
    float* mid0 = (float*)alloc((size_t)Md1 * 64 * 4);  // down1 out [128,32,32,64]
    float* y0   = (float*)alloc(NE * 4);                // down2 out
    float* k1   = (float*)alloc(NE * 4);
    float* k2   = (float*)alloc(NE * 4);
    float* k3   = (float*)alloc(NE * 4);
    float* k4   = (float*)alloc(NE * 4);
    float* k5   = (float*)alloc(NE * 4);
    float* k6   = (float*)alloc(NE * 4);
    float* ytmp = (float*)alloc(NE * 4);
    float* o1   = (float*)alloc(NE * 4);
    float* o2   = (float*)alloc(NE * 4);
    bf16*  A    = (bf16*)alloc((size_t)Md1 * 576 * 2);  // shared im2col buffer (max)
    bf16*  Wc1  = (bf16*)alloc((size_t)64 * 32 * 2);
    bf16*  Wd1c = (bf16*)alloc((size_t)64 * 576 * 2);
    bf16*  Wd1s = (bf16*)alloc((size_t)64 * 64 * 2);
    bf16*  Wd2c = (bf16*)alloc((size_t)64 * 576 * 2);
    bf16*  Wd2s = (bf16*)alloc((size_t)64 * 64 * 2);
    bf16*  Wcc1 = (bf16*)alloc((size_t)64 * 608 * 2);
    bf16*  Wcc2 = (bf16*)alloc((size_t)64 * 608 * 2);
    bf16*  Wden = (bf16*)alloc((size_t)16 * 16384 * 2);
    bf16*  Aden = (bf16*)alloc((size_t)128 * 16384 * 2);
    float* Cden = (float*)alloc((size_t)128 * 16 * 4);
    float* mean = (float*)alloc((size_t)128 * 64 * 4);
    float* rstd = (float*)alloc((size_t)128 * 64 * 4);

    // ---- pack weights to bf16 [Npad, Kpad] ----
    pack(stream, conv1_w, Wc1, 27, 64, 32, 64);
    pack(stream, d1_cw, Wd1c, 576, 64, 576, 64);
    pack(stream, d1_sw, Wd1s, 64, 64, 64, 64);
    pack(stream, d2_cw, Wd2c, 576, 64, 576, 64);
    pack(stream, d2_sw, Wd2s, 64, 64, 64, 64);
    pack(stream, cc1_w, Wcc1, 585, 64, 608, 64);
    pack(stream, cc2_w, Wcc2, 585, 64, 608, 64);
    pack(stream, dw, Wden, 16384, 10, 16384, 16);

    // ---- conv1: 3x3 s1 SAME, 3->64 ----
    im2col(stream, x_in, A, 128, 64, 64, 3, 64, 64, 3, 1, 1, 32, 0, 0.f);
    gemm(stream, A, Wc1, big0, conv1_b, M1, 64, 32, 0);

    // ---- down block 1 (64x64 -> 32x32) ----
    gn_stats<<<128 * 64, 256, 0, stream>>>(big0, mean, rstd, 4096, 64);
    im2col(stream, big0, A, 128, 64, 64, 64, 32, 32, 1, 2, 0, 64, 0, 0.f);   // 1x1 s2 shortcut
    gemm(stream, A, Wd1s, mid0, d1_sb, Md1, 64, 64, 0);
    // GN+relu in place (stats precomputed), then 3x3 s2 conv, accumulate
    gn_apply_launch(stream, big0, big0, mean, rstd, d1_gn_s, d1_gn_b, 128, 4096, 64, 1);
    im2col(stream, big0, A, 128, 64, 64, 64, 32, 32, 3, 2, 0, 576, 0, 0.f);  // SAME s2: pad_before=0
    gemm(stream, A, Wd1c, mid0, d1_cb, Md1, 64, 576, 1);

    // ---- down block 2 (32x32 -> 16x16) ----
    gn_stats<<<128 * 64, 256, 0, stream>>>(mid0, mean, rstd, 1024, 64);
    im2col(stream, mid0, A, 128, 32, 32, 64, 16, 16, 1, 2, 0, 64, 0, 0.f);
    gemm(stream, A, Wd2s, y0, d2_sb, Md2, 64, 64, 0);
    gn_apply_launch(stream, mid0, mid0, mean, rstd, d2_gn_s, d2_gn_b, 128, 1024, 64, 1);
    im2col(stream, mid0, A, 128, 32, 32, 64, 16, 16, 3, 2, 0, 576, 0, 0.f);
    gemm(stream, A, Wd2c, y0, d2_cb, Md2, 64, 576, 1);

    // ---- ODE: one dopri5 step over [0,1] (tol = 1.0 => coarse step) ----
    OdeCtx oc{stream, gn1_s, gn1_b, cc1_b, gn2_s, gn2_b, cc2_b, gn3_s, gn3_b,
              Wcc1, Wcc2, A, o1, o2, mean, rstd};

    ode_f(oc, y0, 0.f, k1);
    wsum(stream, ytmp, NE, y0, 1.f, k1, 0.2f);
    ode_f(oc, ytmp, 0.2f, k2);
    wsum(stream, ytmp, NE, y0, 1.f, k1, 3.f/40.f, k2, 9.f/40.f);
    ode_f(oc, ytmp, 0.3f, k3);
    wsum(stream, ytmp, NE, y0, 1.f, k1, 44.f/45.f, k2, -56.f/15.f, k3, 32.f/9.f);
    ode_f(oc, ytmp, 0.8f, k4);
    wsum(stream, ytmp, NE, y0, 1.f, k1, 19372.f/6561.f, k2, -25360.f/2187.f,
         k3, 64448.f/6561.f, k4, -212.f/729.f);
    ode_f(oc, ytmp, 8.f/9.f, k5);
    wsum(stream, ytmp, NE, y0, 1.f, k1, 9017.f/3168.f, k2, -355.f/33.f,
         k3, 46732.f/5247.f, k4, 49.f/176.f, k5, -5103.f/18656.f);
    ode_f(oc, ytmp, 1.f, k6);
    // 5th-order solution
    wsum(stream, ytmp, NE, y0, 1.f, k1, 35.f/384.f, k3, 500.f/1113.f,
         k4, 125.f/192.f, k5, -2187.f/6784.f, k6, 11.f/84.f);

    // ---- output head: GN->relu, flatten, dense 16384->10, log_softmax ----
    gn(stream, ytmp, o1, gno_s, gno_b, mean, rstd, 128, 256, 64, 1);
    cast_f32_bf16<<<(int)((NE + 255) / 256), 256, 0, stream>>>(o1, Aden, NE);
    gemm(stream, Aden, Wden, Cden, nullptr, 128, 16, 16384, 0);
    logsoftmax10<<<1, 128, 0, stream>>>(Cden, db, (float*)d_out, 128);
}
